// GTS_23716809408871
// MI455X (gfx1250) — compile-verified, matrix operand order checked by reading, not run on hardware
//
#include <hip/hip_runtime.h>
#include <hip/hip_fp16.h>

// ---- problem dims (fixed by reference) ----
#define B_    16
#define P_    12
#define Q_    12
#define N_    512
#define CIN_  2
#define H_    64
#define HP_   32
#define FEMB_ 64
#define FNODE_ 2016
#define TEMP_INV 2.0f   // 1/0.5
#define CATLD 384       // fixed catH row stride (f16), = max Kpad

typedef __attribute__((ext_vector_type(16))) _Float16 v16h;
typedef __attribute__((ext_vector_type(8)))  _Float16 v8h;
typedef __attribute__((ext_vector_type(8)))  float    v8f;

#define LDS_STRIDE 40   // f16 per row: 32 data + 8 pad (16B align kept, banks staggered)

__device__ __forceinline__ v16h frag16(v8h lo, v8h hi) {
    return __builtin_shufflevector(lo, hi, 0,1,2,3,4,5,6,7,8,9,10,11,12,13,14,15);
}

// ---------------------------------------------------------------------------
// Strided-batched f16 GEMM on the CDNA5 matrix core.
//   out = act( alpha * A @ B + beta * Cadd + bias ),  written to f32 C and/or f16 Ch
// A: (M,K) f16 row-major, M % 32 == 0, K % 32 == 0 (operands pre-padded with 0).
// B: if bT==1, B is pre-transposed (Nc_pad x K) -> direct b128 fragment loads.
//    if bT==0, B is (K,Nc) row-major -> LDS transpose staging per K-step.
// All strides are multiples of 8 halves => every fragment is one aligned 16B run.
// No bounds checks in the hot loop; EXEC is all-ones at the WMMA.
// act: 0=none 1=relu 2=sigmoid 3=tanh
// ---------------------------------------------------------------------------
__global__ __launch_bounds__(128)
void wmma_gemm_kernel(const _Float16* __restrict__ A, int lda, long sA,
                      const _Float16* __restrict__ Bm, int ldb, long sB, int bT,
                      float* __restrict__ C, int ldc, long sC,
                      _Float16* __restrict__ Ch, int ldch, long sCh,
                      const _Float16* __restrict__ Cadd, int ldadd, long sAdd,
                      const float* __restrict__ bias,
                      float alpha, float beta, int act,
                      int M, int Nc, int K)
{
    __shared__ __attribute__((aligned(16))) _Float16 smB[32 * LDS_STRIDE];

    const int tid  = threadIdx.x;
    const int lane = tid & 31;
    const int wave = tid >> 5;
    const int bz  = blockIdx.z;
    const int bm0 = blockIdx.y * 32;
    const int bn0 = blockIdx.x * 32;

    const int lm = lane & 15;          // row (A) / col (B,D) within 16x16 tile
    const int kb = (lane >> 4) << 3;   // K half select: 0 or 8
    const int wm = (wave >> 1) * 16;   // wave sub-tile inside 32x32 block tile
    const int wn = (wave & 1) * 16;

    const _Float16* Arow = A + (long)bz * sA + (long)(bm0 + wm + lm) * lda;

    v8f acc = {};

    if (bT) {
        // ---- both operands direct from global: 4x global_load_b128 + wmma ----
        const _Float16* Brow = Bm + (long)bz * sB + (long)(bn0 + wn + lm) * ldb;
        for (int k0 = 0; k0 < K; k0 += 32) {
            const v8h a0 = *(const v8h*)(Arow + k0 + kb);
            const v8h a1 = *(const v8h*)(Arow + k0 + 16 + kb);
            const v8h b0 = *(const v8h*)(Brow + k0 + kb);
            const v8h b1 = *(const v8h*)(Brow + k0 + 16 + kb);
            acc = __builtin_amdgcn_wmma_f32_16x16x32_f16(
                      false, frag16(a0, a1), false, frag16(b0, b1),
                      (short)0, acc, false, false);
        }
    } else {
        // ---- B row-major: LDS transpose staging (adjacency @ features path) ----
        const _Float16* Bb = Bm + (long)bz * sB;
        const int sr  = tid >> 2;        // 0..31 : B tile row (K)
        const int sc0 = (tid & 3) * 8;   // 0,8,16,24 : first col of 8-elem run
        for (int k0 = 0; k0 < K; k0 += 32) {
            const v8h bpack = *(const v8h*)(Bb + (long)(k0 + sr) * ldb + bn0 + sc0);
            if (k0 + 32 < K)             // global_prefetch_b8 for next K tile
                __builtin_prefetch(Bb + (long)(k0 + 32 + sr) * ldb + bn0 + sc0, 0, 1);
#pragma unroll
            for (int i = 0; i < 8; ++i)
                smB[(sc0 + i) * LDS_STRIDE + sr] = bpack[i];
            __syncthreads();
            const v8h a0 = *(const v8h*)(Arow + k0 + kb);
            const v8h a1 = *(const v8h*)(Arow + k0 + 16 + kb);
            const v8h b0 = *(const v8h*)(&smB[(wn + lm) * LDS_STRIDE + kb]);
            const v8h b1 = *(const v8h*)(&smB[(wn + lm) * LDS_STRIDE + 16 + kb]);
            acc = __builtin_amdgcn_wmma_f32_16x16x32_f16(
                      false, frag16(a0, a1), false, frag16(b0, b1),
                      (short)0, acc, false, false);
            __syncthreads();
        }
    }

    // ---- epilogue (M is always a multiple of 32; only N needs a guard) ----
    const int n = bn0 + wn + lm;
    if (n < Nc) {
#pragma unroll
        for (int v = 0; v < 8; ++v) {
            const int m = bm0 + wm + v + kb;   // D layout: M = v + 8*(lane>>4)
            float r = alpha * acc[v];
            if (Cadd) r += beta * (float)Cadd[(long)bz * sAdd + (long)m * ldadd + n];
            if (bias) r += bias[n];
            if      (act == 1) r = fmaxf(r, 0.f);
            else if (act == 2) r = 1.f / (1.f + __expf(-r));
            else if (act == 3) r = tanhf(r);
            if (C)  C [(long)bz * sC  + (long)m * ldc  + n] = r;
            if (Ch) Ch[(long)bz * sCh + (long)m * ldch + n] = (_Float16)r;
        }
    }
}

// ---------------------------------------------------------------------------
// conversion kernels (run once per launch; negligible cost)
// ---------------------------------------------------------------------------
__global__ void cvt_kernel(const float* __restrict__ src, _Float16* __restrict__ dst, int n)
{
    int i = blockIdx.x * blockDim.x + threadIdx.x;
    if (i < n) dst[i] = (_Float16)src[i];
}

// dst (dRows x dK) = transpose of src (K x Ncols), zero padded
__global__ void cvt_T_kernel(const float* __restrict__ src, int K, int Ncols,
                             _Float16* __restrict__ dst, int dRows, int dK)
{
    int i = blockIdx.x * blockDim.x + threadIdx.x;
    if (i >= dRows * dK) return;
    int k = i % dK, r = i / dK;
    float v = (r < Ncols && k < K) ? src[(long)k * Ncols + r] : 0.f;
    dst[i] = (_Float16)v;
}

// gconv weight: src ((3d) x cols) -> dst (cols x Kpad) transposed, rows remapped
// to padded block offsets {0, da, 2da} matching the catH layout.
__global__ void cvt_w_gconv_T_kernel(const float* __restrict__ src, int d, int da,
                                     int Kpad, int cols, _Float16* __restrict__ dst)
{
    int i = blockIdx.x * blockDim.x + threadIdx.x;
    if (i >= cols * Kpad) return;
    int k = i % Kpad, r = i / Kpad;
    float v = 0.f;
    if (k < 2 * da + d) {
        int b = k / da; if (b > 2) b = 2;
        int off = k - b * da;
        if (off < d) v = src[(long)(b * d + off) * cols + r];
    }
    dst[i] = (_Float16)v;
}

// zero the pad columns of catH for this cell: [d,da), [da+d,2da), [2da+d,Kpad)
__global__ void zero_pads_kernel(_Float16* __restrict__ cat, int d, int da, int Kpad)
{
    int p1 = da - d;
    int padTot = 2 * p1 + (Kpad - 2 * da - d);
    long total = (long)B_ * N_ * padTot;
    long i = (long)blockIdx.x * blockDim.x + threadIdx.x;
    if (i >= total) return;
    int p = (int)(i % padTot);
    long rn = i / padTot;
    int c;
    if      (p <     p1) c = d + p;
    else if (p < 2 * p1) c = da + d + (p - p1);
    else                 c = 2 * da + d + (p - 2 * p1);
    cat[rn * CATLD + c] = (_Float16)0.f;
}

// ---------------------------------------------------------------------------
// Pairwise graph learner -> AdjH (f16)
// ---------------------------------------------------------------------------
__global__ void learner_pairwise_kernel(const float* __restrict__ pa,
                                        const float* __restrict__ pb,
                                        const float* __restrict__ b1,
                                        const float* __restrict__ W2,
                                        const float* __restrict__ b2,
                                        const float* __restrict__ u,
                                        _Float16* __restrict__ AdjH)
{
    int idx = blockIdx.x * blockDim.x + threadIdx.x;
    if (idx >= N_ * N_) return;
    int i = idx / N_, j = idx % N_;
    float l0 = b2[0], l1 = b2[1];
#pragma unroll 8
    for (int k = 0; k < HP_; ++k) {
        float h = pa[i * HP_ + k] + pb[j * HP_ + k] + b1[k];
        h = fmaxf(h, 0.f);
        l0 += h * W2[k * 2 + 0];
        l1 += h * W2[k * 2 + 1];
    }
    float g0 = -__logf(-__logf(u[idx * 2 + 0]));
    float g1 = -__logf(-__logf(u[idx * 2 + 1]));
    float z0 = (l0 + g0) * TEMP_INV;
    float z1 = (l1 + g1) * TEMP_INV;
    float mx = fmaxf(z0, z1);
    float e0 = __expf(z0 - mx), e1 = __expf(z1 - mx);
    AdjH[idx] = (_Float16)(e0 / (e0 + e1));
}

// Build gconv x0 block: catH[b,n,0:d] = concat(inp, (ru? r*h : h))  (f16)
__global__ void build_x0_kernel(const float* __restrict__ inp, long inpSB, int inpSN, int din,
                                const float* __restrict__ h, const float* __restrict__ ru,
                                _Float16* __restrict__ cat, int d)
{
    int idx = blockIdx.x * blockDim.x + threadIdx.x;
    if (idx >= B_ * N_ * d) return;
    int c = idx % d;
    int n = (idx / d) % N_;
    int b = idx / (d * N_);
    float v;
    if (c < din) {
        v = inp[(long)b * inpSB + (long)n * inpSN + c];
    } else {
        int k = c - din;
        v = h[((long)b * N_ + n) * H_ + k];
        if (ru) v *= ru[((long)b * N_ + n) * (2 * H_) + k];   // r = first half
    }
    cat[((long)b * N_ + n) * CATLD + c] = (_Float16)v;
}

// GRU state update: h = u*h + (1-u)*c ; optional f16 copy for WMMA consumers
__global__ void gru_update_kernel(float* __restrict__ h,
                                  const float* __restrict__ ru,
                                  const float* __restrict__ c,
                                  _Float16* __restrict__ h16)
{
    int idx = blockIdx.x * blockDim.x + threadIdx.x;
    if (idx >= B_ * N_ * H_) return;
    int k = idx % H_;
    long bn = idx / H_;
    float ug = ru[bn * (2 * H_) + H_ + k];
    float nh = ug * h[idx] + (1.f - ug) * c[idx];
    h[idx] = nh;
    if (h16) h16[idx] = (_Float16)nh;
}

// Record decoder output for step t and feed it back as next input.
__global__ void scatter_out_kernel(const float* __restrict__ outq,
                                   float* __restrict__ xq,
                                   float* __restrict__ out, int t)
{
    int idx = blockIdx.x * blockDim.x + threadIdx.x;
    if (idx >= B_ * N_) return;
    int n = idx % N_;
    int b = idx / N_;
    float v = outq[idx];
    xq[idx] = v;
    out[((long)b * Q_ + t) * N_ + n] = v;
}

// ---------------------------------------------------------------------------
extern "C" void kernel_launch(void* const* d_in, const int* in_sizes, int n_in,
                              void* d_out, int out_size, void* d_ws, size_t ws_size,
                              hipStream_t stream)
{
    (void)in_sizes; (void)n_in; (void)out_size; (void)ws_size;

    const float* x    = (const float*)d_in[0];
    /* d_in[1] = ycl : unused in eval mode */
    const float* nf   = (const float*)d_in[2];
    const float* ug   = (const float*)d_in[3];
    const float* Wf   = (const float*)d_in[4];
    const float* bf   = (const float*)d_in[5];
    const float* W1a  = (const float*)d_in[6];
    const float* W1b  = (const float*)d_in[7];
    const float* b1   = (const float*)d_in[8];
    const float* W2   = (const float*)d_in[9];
    const float* b2   = (const float*)d_in[10];
    struct CellW { const float *Wru, *bru, *Wc, *bc; };
    CellW enc0{(const float*)d_in[11], (const float*)d_in[12], (const float*)d_in[13], (const float*)d_in[14]};
    CellW enc1{(const float*)d_in[15], (const float*)d_in[16], (const float*)d_in[17], (const float*)d_in[18]};
    CellW dec0{(const float*)d_in[19], (const float*)d_in[20], (const float*)d_in[21], (const float*)d_in[22]};
    CellW dec1{(const float*)d_in[23], (const float*)d_in[24], (const float*)d_in[25], (const float*)d_in[26]};
    const float* Wout = (const float*)d_in[27];
    const float* bout = (const float*)d_in[28];
    float* out = (float*)d_out;

    // ---- workspace carve-out: f32 region then f16 region (16B aligned) ----
    float* wf32 = (float*)d_ws;
    float* pa   = wf32;  wf32 += N_ * HP_;
    float* pb   = wf32;  wf32 += N_ * HP_;
    float* ru   = wf32;  wf32 += (long)B_ * N_ * 2 * H_;
    float* cbuf = wf32;  wf32 += (long)B_ * N_ * H_;
    float* h1   = wf32;  wf32 += (long)B_ * N_ * H_;
    float* h2   = wf32;  wf32 += (long)B_ * N_ * H_;
    float* xq   = wf32;  wf32 += B_ * N_;
    float* outq = wf32;  wf32 += B_ * N_;

    _Float16* wh = (_Float16*)wf32;
    _Float16* nfH   = wh;  wh += (long)N_ * FNODE_;      // 512x2016
    _Float16* AdjH  = wh;  wh += N_ * N_;                // 512x512
    _Float16* featH = wh;  wh += N_ * FEMB_;             // 512x64
    _Float16* catH  = wh;  wh += (long)B_ * N_ * CATLD;  // 8192x384
    _Float16* h2H   = wh;  wh += (long)B_ * N_ * H_;     // 8192x64
    _Float16* WfT   = wh;  wh += FEMB_ * FNODE_;         // 64x2016
    _Float16* W1aT  = wh;  wh += HP_ * FEMB_;            // 32x64
    _Float16* W1bT  = wh;  wh += HP_ * FEMB_;
    _Float16* WoutT = wh;  wh += 32 * H_;                // 32x64 (rows 1..31 zero)
    // gconv weights: [cell][WruT(128 x Kpad) | WcT(64 x Kpad)]
    auto kpad_of = [](int d) { int da = (d + 7) & ~7; return ((2 * da + d) + 31) & ~31; };
    const int dEnc0 = CIN_ + H_, dB = H_ + H_, dDec0 = 1 + H_;
    const int kpE0 = kpad_of(dEnc0), kpB = kpad_of(dB), kpD0 = kpad_of(dDec0);
    _Float16* e0WruT = wh; wh += 2 * H_ * kpE0;  _Float16* e0WcT = wh; wh += H_ * kpE0;
    _Float16* e1WruT = wh; wh += 2 * H_ * kpB;   _Float16* e1WcT = wh; wh += H_ * kpB;
    _Float16* d0WruT = wh; wh += 2 * H_ * kpD0;  _Float16* d0WcT = wh; wh += H_ * kpD0;
    _Float16* d1WruT = wh; wh += 2 * H_ * kpB;   _Float16* d1WcT = wh; wh += H_ * kpB;

    auto gemm = [&](const _Float16* A, int lda, long sA,
                    const _Float16* Bm, int ldb, long sB, int bT,
                    float* C, int ldc, long sC,
                    _Float16* Ch, int ldch, long sCh,
                    const _Float16* Cadd, int ldadd, long sAdd,
                    const float* bias, float alpha, float beta, int act,
                    int M, int Nc, int K, int batch) {
        dim3 grid((Nc + 31) / 32, (M + 31) / 32, batch);
        wmma_gemm_kernel<<<grid, 128, 0, stream>>>(A, lda, sA, Bm, ldb, sB, bT,
                                                   C, ldc, sC, Ch, ldch, sCh,
                                                   Cadd, ldadd, sAdd,
                                                   bias, alpha, beta, act, M, Nc, K);
    };
    auto ew = [&](int n) { return dim3((n + 255) / 256); };

    // ===== one-time operand conversion =====
    cvt_kernel<<<ew(N_ * FNODE_), 256, 0, stream>>>(nf, nfH, N_ * FNODE_);
    cvt_T_kernel<<<ew(FEMB_ * FNODE_), 256, 0, stream>>>(Wf, FNODE_, FEMB_, WfT, FEMB_, FNODE_);
    cvt_T_kernel<<<ew(HP_ * FEMB_), 256, 0, stream>>>(W1a, FEMB_, HP_, W1aT, HP_, FEMB_);
    cvt_T_kernel<<<ew(HP_ * FEMB_), 256, 0, stream>>>(W1b, FEMB_, HP_, W1bT, HP_, FEMB_);
    cvt_T_kernel<<<ew(32 * H_), 256, 0, stream>>>(Wout, H_, 1, WoutT, 32, H_);
    struct GW { const CellW* w; int d; int kp; _Float16 *ruT, *cT; };
    GW gws[4] = {{&enc0, dEnc0, kpE0, e0WruT, e0WcT}, {&enc1, dB, kpB, e1WruT, e1WcT},
                 {&dec0, dDec0, kpD0, d0WruT, d0WcT}, {&dec1, dB, kpB, d1WruT, d1WcT}};
    for (auto& g : gws) {
        int da = (g.d + 7) & ~7;
        cvt_w_gconv_T_kernel<<<ew(2 * H_ * g.kp), 256, 0, stream>>>(g.w->Wru, g.d, da, g.kp, 2 * H_, g.ruT);
        cvt_w_gconv_T_kernel<<<ew(H_ * g.kp), 256, 0, stream>>>(g.w->Wc, g.d, da, g.kp, H_, g.cT);
    }

    // one DCGRU cell: updates h (f32) in place; h16 optionally mirrored.
    auto cell = [&](const float* inp, long inpSB, int inpSN, int din,
                    float* h, const CellW& cw, const _Float16* WruT, const _Float16* WcT,
                    _Float16* h16) {
        const int d = din + H_;
        const int da = (d + 7) & ~7;
        const int Kp = kpad_of(d);
        const long bstr = (long)N_ * CATLD;
        const int totX = B_ * N_ * d;
        const int padTot = 2 * (da - d) + (Kp - 2 * da - d);
        // gate path: catH = [x0 | A@x0 | 2A@x1-x0] (blocks at 0, da, 2da)
        build_x0_kernel<<<ew(totX), 256, 0, stream>>>(inp, inpSB, inpSN, din, h, nullptr, catH, d);
        if (padTot > 0)
            zero_pads_kernel<<<ew(B_ * N_ * padTot), 256, 0, stream>>>(catH, d, da, Kp);
        gemm(AdjH, N_, 0, catH, CATLD, bstr, 0, nullptr, 0, 0,
             catH + da, CATLD, bstr, nullptr, 0, 0,
             nullptr, 1.f, 0.f, 0, N_, d, N_, B_);
        gemm(AdjH, N_, 0, catH + da, CATLD, bstr, 0, nullptr, 0, 0,
             catH + 2 * da, CATLD, bstr, catH, CATLD, bstr,
             nullptr, 2.f, -1.f, 0, N_, d, N_, B_);
        gemm(catH, CATLD, 0, WruT, Kp, 0, 1, ru, 2 * H_, 0, nullptr, 0, 0,
             nullptr, 0, 0, cw.bru, 1.f, 0.f, /*sigmoid*/2, B_ * N_, 2 * H_, Kp, 1);
        // candidate path: x0' = [inp | r*h]
        build_x0_kernel<<<ew(totX), 256, 0, stream>>>(inp, inpSB, inpSN, din, h, ru, catH, d);
        gemm(AdjH, N_, 0, catH, CATLD, bstr, 0, nullptr, 0, 0,
             catH + da, CATLD, bstr, nullptr, 0, 0,
             nullptr, 1.f, 0.f, 0, N_, d, N_, B_);
        gemm(AdjH, N_, 0, catH + da, CATLD, bstr, 0, nullptr, 0, 0,
             catH + 2 * da, CATLD, bstr, catH, CATLD, bstr,
             nullptr, 2.f, -1.f, 0, N_, d, N_, B_);
        gemm(catH, CATLD, 0, WcT, Kp, 0, 1, cbuf, H_, 0, nullptr, 0, 0,
             nullptr, 0, 0, cw.bc, 1.f, 0.f, /*tanh*/3, B_ * N_, H_, Kp, 1);
        gru_update_kernel<<<ew(B_ * N_ * H_), 256, 0, stream>>>(h, ru, cbuf, h16);
    };

    // ===== graph structure learner =====
    gemm(nfH, FNODE_, 0, WfT, FNODE_, 0, 1, nullptr, 0, 0, featH, FEMB_, 0,
         nullptr, 0, 0, bf, 1.f, 0.f, /*relu*/1, N_, FEMB_, FNODE_, 1);
    gemm(featH, FEMB_, 0, W1aT, FEMB_, 0, 1, pa, HP_, 0, nullptr, 0, 0,
         nullptr, 0, 0, nullptr, 1.f, 0.f, 0, N_, HP_, FEMB_, 1);
    gemm(featH, FEMB_, 0, W1bT, FEMB_, 0, 1, pb, HP_, 0, nullptr, 0, 0,
         nullptr, 0, 0, nullptr, 1.f, 0.f, 0, N_, HP_, FEMB_, 1);
    learner_pairwise_kernel<<<ew(N_ * N_), 256, 0, stream>>>(pa, pb, b1, W2, b2, ug, AdjH);

    // ===== encoder =====
    hipMemsetAsync(h1, 0, (size_t)B_ * N_ * H_ * sizeof(float), stream);
    hipMemsetAsync(h2, 0, (size_t)B_ * N_ * H_ * sizeof(float), stream);
    for (int t = 0; t < P_; ++t) {
        cell(x + (long)t * N_ * CIN_, (long)P_ * N_ * CIN_, CIN_, CIN_, h1, enc0, e0WruT, e0WcT, nullptr);
        cell(h1, (long)N_ * H_, H_, H_, h2, enc1, e1WruT, e1WcT, nullptr);
    }

    // ===== decoder (hidden init = encoder finals in h1/h2) =====
    hipMemsetAsync(xq, 0, (size_t)B_ * N_ * sizeof(float), stream);
    for (int t = 0; t < Q_; ++t) {
        cell(xq, (long)N_, 1, 1, h1, dec0, d0WruT, d0WcT, nullptr);
        cell(h1, (long)N_ * H_, H_, H_, h2, dec1, d1WruT, d1WcT, h2H);
        gemm(h2H, H_, 0, WoutT, H_, 0, 1, outq, 1, 0, nullptr, 0, 0,
             nullptr, 0, 0, bout, 1.f, 0.f, 0, B_ * N_, 1, H_, 1);
        scatter_out_kernel<<<ew(B_ * N_), 256, 0, stream>>>(outq, xq, out, t);
    }
}
